// U_MLP_11501922418777
// MI455X (gfx1250) — matrix operand order repaired and checked by reading, not compiled, not run on hardware
//
#include <hip/hip_runtime.h>
#include <math.h>

#define B_   1024
#define S_   60
#define D_   256
#define H_   1024
#define E_   8
#define CAP_ 256
#define SD_  (S_ * D_)   // 15360

typedef __attribute__((ext_vector_type(16))) __bf16 v16bf;
typedef __attribute__((ext_vector_type(8)))  float  v8f;
typedef unsigned short u16;
typedef unsigned int   u32;

// LDS partition (elements of u16); all offsets 16B-aligned
#define XS_OFF   0                    // 64 x 264
#define W1_OFF   (64 * 264)           // 32 x 264
#define W2_OFF   (W1_OFF + 32 * 264)  // 256 x 40
#define HS_OFF   (W2_OFF + 256 * 40)  // 64 x 40
#define SMEM_U16 (HS_OFF + 64 * 40)   // 38144 u16 = 76288 B

// ---------- helpers ----------

__device__ __forceinline__ u16 f2bf(float f) {
  u32 u = __float_as_uint(f);
  u32 r = u + 0x7FFFu + ((u >> 16) & 1u);   // round-to-nearest-even
  return (u16)(r >> 16);
}

struct alignas(16) Q16 { u32 a, b, c, d; };
union FragU { Q16 q[2]; v16bf v; };

// A-matrix 16x32 bf16 layout (ISA 7.12.2): lane m=l&15; lanes 0-15 hold
// K = [0..7] & [16..23], lanes 16-31 hold K = [8..15] & [24..31].
__device__ __forceinline__ v16bf load_frag_a(const u16* buf, int stride,
                                             int row, int k0, int lane) {
  const int koff = (lane >> 4) * 8;
  const u16* p = buf + row * stride + k0 + koff;
  FragU f;
  f.q[0] = *(const Q16*)(p);
  f.q[1] = *(const Q16*)(p + 16);
  return f.v;
}

// B-matrix 32x16 bf16 layout: lane n=l&15; lanes 0-15 hold K = [0..7] & [16..23],
// lanes 16-31 hold K = [8..15] & [24..31] (matches A, K-transposed storage).
__device__ __forceinline__ v16bf load_frag_b(const u16* buf, int stride,
                                             int row, int k0, int lane) {
  const int koff = (lane >> 4) * 8;
  const u16* p = buf + row * stride + k0 + koff;
  FragU f;
  f.q[0] = *(const Q16*)(p);
  f.q[1] = *(const Q16*)(p + 16);
  return f.v;
}

// CDNA5 async copy global->LDS (VFLAT GLOBAL_LOAD_ASYNC_TO_LDS_B128, GV mode).
__device__ __forceinline__ void async_copy_b128(u32 ldsaddr, const void* gptr) {
  asm volatile("global_load_async_to_lds_b128 %0, %1, off"
               :: "v"(ldsaddr), "v"(gptr) : "memory");
}
#define WAIT_ASYNCCNT(imm) asm volatile("s_wait_asynccnt " imm ::: "memory")

__device__ __forceinline__ u32 lds_addr(const void* p) {
  return (u32)(size_t)p;     // flat shared pointer: LDS offset = addr[31:0]
}

// ---------- 0a) weight transpose + bf16 convert pre-pass ----------
// src: [E][R][C] fp32  ->  dst: [E][C][R] bf16

__global__ __launch_bounds__(256)
void transpose_bf16_kernel(const float* __restrict__ src, u16* __restrict__ dst,
                           int R, int C) {
  const int e = blockIdx.z;
  const int c0 = blockIdx.x * 32, r0 = blockIdx.y * 32;
  __shared__ float tile[32][33];
  const int tx = threadIdx.x & 31, ty = threadIdx.x >> 5;   // 32x8
  const float* s = src + (size_t)e * R * C;
  u16* d = dst + (size_t)e * R * C;
#pragma unroll
  for (int i = 0; i < 4; ++i) {
    const int r = r0 + ty + i * 8;
    tile[ty + i * 8][tx] = s[(size_t)r * C + c0 + tx];
  }
  __syncthreads();
#pragma unroll
  for (int i = 0; i < 4; ++i) {
    const int c = c0 + ty + i * 8;
    d[(size_t)c * R + r0 + tx] = f2bf(tile[tx][ty + i * 8]);
  }
}

// ---------- 0b) x -> bf16 convert pre-pass ----------

__global__ __launch_bounds__(256)
void convert_x_kernel(const float* __restrict__ x, u16* __restrict__ xbf) {
  const size_t i = ((size_t)blockIdx.x * 256 + threadIdx.x) * 4;
  const float4 v = *(const float4*)(x + i);
  const u32 lo = (u32)f2bf(v.x) | ((u32)f2bf(v.y) << 16);
  const u32 hi = (u32)f2bf(v.z) | ((u32)f2bf(v.w) << 16);
  *(u32*)(xbf + i)     = lo;
  *(u32*)(xbf + i + 2) = hi;
}

// ---------- 1) routing: logits + softmax ----------

__global__ __launch_bounds__(256)
void routing_kernel(const float* __restrict__ x, const float* __restrict__ Wsw,
                    const float* __restrict__ bsw, float* __restrict__ probs) {
  const int b = blockIdx.x, t = threadIdx.x;
  float a[E_];
#pragma unroll
  for (int e = 0; e < E_; ++e) a[e] = 0.0f;
  const float* xr = x + (size_t)b * SD_;
  for (int i = t; i < SD_; i += 256) {
    const float xv = xr[i];
    const float* w = Wsw + (size_t)i * E_;
#pragma unroll
    for (int e = 0; e < E_; ++e) a[e] += xv * w[e];
  }
  __shared__ float red[256][E_];
#pragma unroll
  for (int e = 0; e < E_; ++e) red[t][e] = a[e];
  __syncthreads();
  for (int off = 128; off > 0; off >>= 1) {
    if (t < off) {
#pragma unroll
      for (int e = 0; e < E_; ++e) red[t][e] += red[t + off][e];
    }
    __syncthreads();
  }
  if (t == 0) {
    float l[E_], mx = -1e30f;
#pragma unroll
    for (int e = 0; e < E_; ++e) { l[e] = red[0][e] + bsw[e]; mx = fmaxf(mx, l[e]); }
    float sum = 0.0f;
#pragma unroll
    for (int e = 0; e < E_; ++e) { l[e] = __expf(l[e] - mx); sum += l[e]; }
    const float inv = 1.0f / sum;
#pragma unroll
    for (int e = 0; e < E_; ++e) probs[(size_t)b * E_ + e] = l[e] * inv;
  }
}

// ---------- 2) per-expert top-CAP via bitonic sort ----------

__global__ __launch_bounds__(512)
void topk_kernel(const float* __restrict__ probs, int* __restrict__ idxb) {
  const int e = blockIdx.x, t = threadIdx.x;
  __shared__ float key[B_];
  __shared__ int   val[B_];
  for (int i = t; i < B_; i += 512) { key[i] = probs[(size_t)i * E_ + e]; val[i] = i; }
  __syncthreads();
  for (int k = 2; k <= B_; k <<= 1) {
    for (int j = k >> 1; j > 0; j >>= 1) {
      for (int i = t; i < B_; i += 512) {
        const int ixj = i ^ j;
        if (ixj > i) {
          const float ka = key[i], kb = key[ixj];
          const int va = val[i], vb = val[ixj];
          const bool a_first = (ka > kb) || (ka == kb && va < vb);
          const bool desc = ((i & k) == 0);
          if (desc ? !a_first : a_first) {
            key[i] = kb; key[ixj] = ka; val[i] = vb; val[ixj] = va;
          }
        }
      }
      __syncthreads();
    }
  }
  if (t < CAP_) idxb[e * CAP_ + t] = val[t];
}

// ---------- 3) fused grouped expert MLP: WMMA + pipelined async DMA ----------
// Per block: 64 gathered rows, full D=256 output, loop H in chunks of 32.
// Async loads complete in order, so s_wait_asynccnt 4 retires the older chunk
// while the newer one stays in flight -> DMA overlaps WMMA.

__global__ __launch_bounds__(256)
void expert_mlp_kernel(const u16* __restrict__ xbf,   // [B][S][D] bf16
                       const u16* __restrict__ w1t,   // [E][H][D] bf16
                       const float* __restrict__ b1,
                       const u16* __restrict__ w2t,   // [E][D][H] bf16
                       const float* __restrict__ b2,
                       const int* __restrict__ idxb,
                       float* __restrict__ accb) {
  const int e = blockIdx.y;
  const int rb = blockIdx.x;            // 0..239 (64 rows each of 15360)
  const int tid = threadIdx.x;
  const int wave = tid >> 5, lane = tid & 31;

  extern __shared__ u16 smem[];
  u16* xs    = smem + XS_OFF;           // 64 x 264 x tile
  u16* w1buf = smem + W1_OFF;           // 32 x 264 W1^T chunk
  u16* w2buf = smem + W2_OFF;           // 256 x 40 W2^T chunk
  u16* hsh   = smem + HS_OFF;           // 64 x 40  gelu(h) chunk

  const u16* w1te = w1t + (size_t)e * H_ * D_;
  const u16* w2te = w2t + (size_t)e * D_ * H_;

  // --- prologue: issue all async DMA (xs: 8/thread, W1: 4, W2: 4) ---
  {
    const int lane8 = lane << 3;
#pragma unroll
    for (int q = 0; q < 8; ++q) {
      const int j = q * 8 + wave;
      const int g = rb * 64 + j;
      const int c = g / S_, s = g - c * S_;
      const int b = idxb[e * CAP_ + c];
      async_copy_b128(lds_addr(&xs[j * 264 + lane8]),
                      xbf + ((size_t)(b * S_ + s)) * D_ + lane8);
    }
  }
#pragma unroll
  for (int q = 0; q < 4; ++q) {         // W1 chunk 0: 32 rows x 512B
    const int cc = tid + q * 256;
    const int n = cc >> 5, off = (cc & 31) << 3;
    async_copy_b128(lds_addr(&w1buf[n * 264 + off]),
                    w1te + (size_t)n * D_ + off);
  }
#pragma unroll
  for (int q = 0; q < 4; ++q) {         // W2 chunk 0: 256 rows x 64B
    const int cc = tid + q * 256;
    const int n = cc >> 2, off = (cc & 3) << 3;
    async_copy_b128(lds_addr(&w2buf[n * 40 + off]),
                    w2te + (size_t)n * H_ + off);
  }

  v8f acc[8];
#pragma unroll
  for (int j = 0; j < 8; ++j)
#pragma unroll
    for (int v = 0; v < 8; ++v) acc[j][v] = 0.0f;

  for (int hc = 0; hc < H_; hc += 32) {
    WAIT_ASYNCCNT("0x4");               // xs + W1(i) landed; W2(i) may be in flight
    __syncthreads();                    // publish xs/w1buf

    // GEMM1: h[64x32] = xs[64x256] @ W1chunk — one 16x16 tile per wave
    {
      const int mt = wave >> 1, nt = wave & 1;
      v8f c1;
#pragma unroll
      for (int v = 0; v < 8; ++v) c1[v] = 0.0f;
#pragma unroll
      for (int kk = 0; kk < 8; ++kk) {
        const v16bf a = load_frag_a(xs,    264, mt * 16 + (lane & 15), kk * 32, lane);
        const v16bf b = load_frag_b(w1buf, 264, nt * 16 + (lane & 15), kk * 32, lane);
        c1 = __builtin_amdgcn_wmma_f32_16x16x32_bf16(false, a, false, b,
                                                     (short)0, c1, false, false);
      }
      const int n = nt * 16 + (lane & 15);
      const float bias = b1[(size_t)e * H_ + hc + n];
#pragma unroll
      for (int v = 0; v < 8; ++v) {
        const int m = mt * 16 + v + ((lane >> 4) << 3);
        float hv = c1[v] + bias;
        hv = 0.5f * hv * (1.0f + erff(hv * 0.70710678118f));   // exact gelu
        hsh[m * 40 + n] = f2bf(hv);
      }
    }
    __syncthreads();                    // w1buf free, hsh published

    if (hc + 32 < H_) {
      // issue W1(i+1) DMA -> overlaps GEMM2(i)
#pragma unroll
      for (int q = 0; q < 4; ++q) {
        const int cc = tid + q * 256;
        const int n = cc >> 5, off = (cc & 31) << 3;
        async_copy_b128(lds_addr(&w1buf[n * 264 + off]),
                        w1te + (size_t)(hc + 32 + n) * D_ + off);
      }
      if (hc + 64 < H_)                 // warm L2 two chunks ahead
        __builtin_prefetch(w1te + (size_t)(hc + 64 + (tid >> 3)) * D_ + (tid & 7) * 32, 0, 0);
      WAIT_ASYNCCNT("0x4");             // W2(i) landed; W1(i+1) in flight
    } else {
      WAIT_ASYNCCNT("0x0");             // final chunk: drain everything
    }
    __syncthreads();                    // publish w2buf

    // GEMM2: acc[64x256] += h[64x32] @ W2chunk — 8 tiles per wave, K=32
    {
      const int mt = wave & 3, ntb = (wave >> 2) * 8;
      const v16bf a = load_frag_a(hsh, 40, mt * 16 + (lane & 15), 0, lane);
      v16bf bfr[8];
#pragma unroll
      for (int j = 0; j < 8; ++j)
        bfr[j] = load_frag_b(w2buf, 40, (ntb + j) * 16 + (lane & 15), 0, lane);
#pragma unroll
      for (int j = 0; j < 8; ++j)
        acc[j] = __builtin_amdgcn_wmma_f32_16x16x32_bf16(false, a, false, bfr[j],
                                                         (short)0, acc[j], false, false);
    }
    __syncthreads();                    // w2buf free

    if (hc + 32 < H_) {
      // issue W2(i+1) DMA -> overlaps GEMM1(i+1)
#pragma unroll
      for (int q = 0; q < 4; ++q) {
        const int cc = tid + q * 256;
        const int n = cc >> 2, off = (cc & 3) << 3;
        async_copy_b128(lds_addr(&w2buf[n * 40 + off]),
                        w2te + (size_t)n * H_ + hc + 32 + off);
      }
      if (hc + 64 < H_)
        __builtin_prefetch(w2te + (size_t)tid * H_ + hc + 64, 0, 0);
    }
  }

  // epilogue: bias + atomic scatter-add into fp32 accumulator
  {
    const int mt = wave & 3, ntb = (wave >> 2) * 8;
#pragma unroll
    for (int v = 0; v < 8; ++v) {
      const int m = mt * 16 + v + ((lane >> 4) << 3);
      const int g = rb * 64 + m;
      const int c = g / S_, s = g - c * S_;
      const int b = idxb[e * CAP_ + c];
      float* dst = accb + ((size_t)(b * S_ + s)) * D_;
#pragma unroll
      for (int j = 0; j < 8; ++j) {
        const int n = (ntb + j) * 16 + (lane & 15);
        atomicAdd(dst + n, acc[j][v] + b2[(size_t)e * D_ + n]);
      }
    }
  }
}

// ---------- 4) residual + LayerNorm ----------

__global__ __launch_bounds__(256)
void ln_kernel(const float* __restrict__ x, const float* __restrict__ acc,
               const float* __restrict__ gamma, const float* __restrict__ beta,
               float* __restrict__ out) {
  const size_t row = blockIdx.x;
  const int t = threadIdx.x;
  const size_t off = row * D_ + t;
  const float z = x[off] + acc[off];
  __shared__ float s1[256], s2[256];
  s1[t] = z; s2[t] = z * z;
  __syncthreads();
  for (int o = 128; o > 0; o >>= 1) {
    if (t < o) { s1[t] += s1[t + o]; s2[t] += s2[t + o]; }
    __syncthreads();
  }
  const float mu = s1[0] * (1.0f / D_);
  const float var = s2[0] * (1.0f / D_) - mu * mu;
  const float r = rsqrtf(var + 1e-5f);
  out[off] = (z - mu) * r * gamma[t] + beta[t];
}

// ---------- launcher ----------

extern "C" void kernel_launch(void* const* d_in, const int* in_sizes, int n_in,
                              void* d_out, int out_size, void* d_ws, size_t ws_size,
                              hipStream_t stream) {
  (void)in_sizes; (void)n_in; (void)out_size; (void)ws_size;
  const float* x     = (const float*)d_in[0];
  const float* Wsw   = (const float*)d_in[1];
  const float* bsw   = (const float*)d_in[2];
  const float* W1    = (const float*)d_in[3];
  const float* b1    = (const float*)d_in[4];
  const float* W2    = (const float*)d_in[5];
  const float* b2    = (const float*)d_in[6];
  const float* gamma = (const float*)d_in[7];
  const float* beta  = (const float*)d_in[8];
  float* out = (float*)d_out;

  float* acc   = (float*)d_ws;                           // B*S*D fp32 accumulator
  float* probs = acc + (size_t)B_ * S_ * D_;             // B*E
  int*   idxb  = (int*)(probs + (size_t)B_ * E_);        // E*CAP
  u16*   w1t   = (u16*)(idxb + (size_t)E_ * CAP_);       // [E][H][D] bf16
  u16*   w2t   = w1t + (size_t)E_ * H_ * D_;             // [E][D][H] bf16
  u16*   xbf   = w2t + (size_t)E_ * D_ * H_;             // [B][S][D] bf16

  hipMemsetAsync(acc, 0, sizeof(float) * (size_t)B_ * S_ * D_, stream);

  // pre-convert operands to bf16 (weights transposed) for async DMA staging
  transpose_bf16_kernel<<<dim3(H_ / 32, D_ / 32, E_), 256, 0, stream>>>(W1, w1t, D_, H_);
  transpose_bf16_kernel<<<dim3(D_ / 32, H_ / 32, E_), 256, 0, stream>>>(W2, w2t, H_, D_);
  convert_x_kernel<<<(B_ * S_ * D_) / 1024, 256, 0, stream>>>(x, xbf);

  routing_kernel<<<B_, 256, 0, stream>>>(x, Wsw, bsw, probs);
  topk_kernel<<<E_, 512, 0, stream>>>(probs, idxb);
  expert_mlp_kernel<<<dim3(240, E_), 256, SMEM_U16 * sizeof(u16), stream>>>(
      xbf, w1t, b1, w2t, b2, idxb, acc);
  ln_kernel<<<B_ * S_, 256, 0, stream>>>(x, acc, gamma, beta, out);
}
